// GCN_80272938762672
// MI455X (gfx1250) — matrix operand order repaired and checked by reading, not compiled, not run on hardware
//
#include <hip/hip_runtime.h>
#include <hip/hip_bf16.h>

// ---------- types ----------
typedef __attribute__((ext_vector_type(16))) _Float16 v16h;
typedef __attribute__((ext_vector_type(8)))  float    v8f;
typedef __attribute__((ext_vector_type(4)))  unsigned u32x4;
typedef __attribute__((ext_vector_type(2)))  unsigned u32x2;
typedef __attribute__((ext_vector_type(4)))  float    f32x4;

#define NNODES 100000
#define NEDGES 300000
#define NGRAPH 1024
#define EMB    256

// ---------- elementwise / setup kernels ----------
__global__ __launch_bounds__(256) void k_zero_f32(float* p, int n) {
  int t = blockIdx.x * 256 + threadIdx.x;
  if (t < n) p[t] = 0.0f;
}

__global__ __launch_bounds__(256) void k_deg_scatter(const int* __restrict__ dst,
                                                     float* __restrict__ deg, int e) {
  int t = blockIdx.x * 256 + threadIdx.x;
  if (t < e) atomicAdd(&deg[dst[t]], 1.0f);
}

__global__ __launch_bounds__(256) void k_deg_to_dinv(float* __restrict__ d, int n) {
  int t = blockIdx.x * 256 + threadIdx.x;
  if (t < n) d[t] = rsqrtf(d[t] + 1.0f);
}

// layer-0 dense: hW[i][f] = sum_k x[i][k] * W0[k][f], K=9
__global__ __launch_bounds__(256) void k_gemm0(const float* __restrict__ x,
                                               const float* __restrict__ W0,
                                               float* __restrict__ hW) {
  __shared__ float xs[9];
  int i = blockIdx.x;
  int f = threadIdx.x;
  if (f < 9) xs[f] = x[(size_t)i * 9 + f];
  __syncthreads();
  float acc = 0.0f;
#pragma unroll
  for (int k = 0; k < 9; ++k) acc = fmaf(xs[k], W0[k * EMB + f], acc);
  hW[(size_t)i * EMB + f] = acc;
}

// transpose + convert weights: W f32 [K][N] -> Wt f16 [N][K]
__global__ __launch_bounds__(256) void k_wcvt_t(const float* __restrict__ W,
                                                _Float16* __restrict__ Wt,
                                                int K, int N) {
  int t = blockIdx.x * 256 + threadIdx.x;
  if (t >= K * N) return;
  int n = t / K, k = t - n * K;
  Wt[(size_t)n * K + k] = (_Float16)W[(size_t)k * N + n];
}

// agg[i][f] = hW[i][f]*dinv[i]^2 + b[f]   (self-loop + bias), float4 per thread
__global__ __launch_bounds__(256) void k_agg_init(const float* __restrict__ hW,
                                                  const float* __restrict__ dinv,
                                                  const float* __restrict__ bias,
                                                  float* __restrict__ agg, int n4) {
  int t = blockIdx.x * 256 + threadIdx.x;
  if (t >= n4) return;
  int i = t >> 6, g = t & 63;
  float s = dinv[i]; s = s * s;
  f32x4 hv = *(const f32x4*)(hW + (size_t)i * EMB + g * 4);
  f32x4 b4 = *(const f32x4*)(bias + g * 4);
  f32x4 r;
  r.x = fmaf(hv.x, s, b4.x); r.y = fmaf(hv.y, s, b4.y);
  r.z = fmaf(hv.z, s, b4.z); r.w = fmaf(hv.w, s, b4.w);
  *(f32x4*)(agg + (size_t)i * EMB + g * 4) = r;
}

// per-edge scatter: agg[dst] += hW[src] * dinv[src]*dinv[dst]
__global__ __launch_bounds__(256) void k_edge_scatter(const int* __restrict__ src,
                                                      const int* __restrict__ dst,
                                                      const float* __restrict__ hW,
                                                      const float* __restrict__ dinv,
                                                      float* __restrict__ agg, int e64) {
  int t = blockIdx.x * 256 + threadIdx.x;
  if (t >= e64) return;
  int e = t >> 6, g = t & 63;
  int s = src[e], d = dst[e];
  float w = dinv[s] * dinv[d];
  f32x4 hv = *(const f32x4*)(hW + (size_t)s * EMB + g * 4);
  float* ap = agg + (size_t)d * EMB + g * 4;
  atomicAdd(ap + 0, hv.x * w);
  atomicAdd(ap + 1, hv.y * w);
  atomicAdd(ap + 2, hv.z * w);
  atomicAdd(ap + 3, hv.w * w);
}

// relu + f32->f16: h16 = max(agg,0)
__global__ __launch_bounds__(256) void k_relu_cvt(const float* __restrict__ agg,
                                                  _Float16* __restrict__ h16, int n4) {
  int t = blockIdx.x * 256 + threadIdx.x;
  if (t >= n4) return;
  f32x4 v = *(const f32x4*)(agg + (size_t)t * 4);
  union { _Float16 h[4]; u32x2 u; } p;
  p.h[0] = (_Float16)fmaxf(v.x, 0.0f);
  p.h[1] = (_Float16)fmaxf(v.y, 0.0f);
  p.h[2] = (_Float16)fmaxf(v.z, 0.0f);
  p.h[3] = (_Float16)fmaxf(v.w, 0.0f);
  *(u32x2*)(h16 + (size_t)t * 4) = p.u;
}

// ---------- async global->LDS staging (CDNA5 async path, ASYNCcnt) ----------
// Stages 64B per thread: INST_OFFSET is added to BOTH the LDS and the global
// address (ISA 15.18.3), so one address pair + 4 offsets covers the panel slice.
__device__ __forceinline__ void async_copy64(unsigned ldsoff, const _Float16* g) {
  unsigned long long ga = (unsigned long long)g;
  asm volatile(
      "global_load_async_to_lds_b128 %0, %1, off\n\t"
      "global_load_async_to_lds_b128 %0, %1, off offset:16\n\t"
      "global_load_async_to_lds_b128 %0, %1, off offset:32\n\t"
      "global_load_async_to_lds_b128 %0, %1, off offset:48"
      :
      : "v"(ldsoff), "v"(ga)
      : "memory");
}

__device__ __forceinline__ void wait_async0() {
  asm volatile("s_wait_asynccnt 0x0" ::: "memory");
}

// ---------- WMMA GEMM: C[M][Nstride] = A_f16[M][K] x Wt_f16[N][K]^T ----------
// block = 256 threads = 8 waves; wave w owns rows m0..m0+15; block covers 128 rows x 256 cols.
// B panel (16KB) is double-buffered in LDS and filled by async DMA loads that
// overlap the 16 WMMAs of the previous k-step. B fragments are additionally
// double-buffered in registers so each WMMA only waits for LDS loads issued
// one WMMA earlier (latency hidden behind the matrix op).
__global__ __launch_bounds__(256) void k_gemm_wmma(const _Float16* __restrict__ A,
                                                   const _Float16* __restrict__ Wt,
                                                   float* __restrict__ C,
                                                   int M, int Nstride, int K) {
  __shared__ _Float16 panel[2][256 * 32];  // [buf][n_local][k_local] f16, 2 x 16 KB
  const int tid  = threadIdx.x;
  const int wave = tid >> 5;
  const int lane = tid & 31;
  const int l15  = lane & 15;
  const int lhi  = lane >> 4;
  const int m0   = blockIdx.x * 128 + wave * 16;
  const int n0   = blockIdx.y * 256;

  // generic LDS pointers: low 32 bits are the LDS byte offset (flat aperture)
  const unsigned lds0 = (unsigned)(size_t)(&panel[0][tid * 32]);
  const unsigned lds1 = (unsigned)(size_t)(&panel[1][tid * 32]);
  const _Float16* wrow = Wt + (size_t)(n0 + tid) * K;

  v8f acc[16];
#pragma unroll
  for (int t = 0; t < 16; ++t)
#pragma unroll
    for (int j = 0; j < 8; ++j) acc[t][j] = 0.0f;

  int arow = m0 + l15; if (arow > M - 1) arow = M - 1;  // clamp (stores guarded)
  const int ksteps = K >> 5;

  // prologue: fill buffer 0
  async_copy64(lds0, wrow);

  for (int kk = 0; kk < ksteps; ++kk) {
    const int cur = kk & 1;
    wait_async0();        // this wave's panel slice landed in LDS
    __syncthreads();      // whole panel ready; prior reads of the other buffer done
    if (kk + 1 < ksteps)  // prefetch next panel into the other buffer (overlaps WMMAs)
      async_copy64(cur ? lds0 : lds1, wrow + (kk + 1) * 32);

    // A fragment (ISA 16-bit 16x32 layout): lanes 0-15: K 0..7 & 16..23; lanes 16-31: K 8..15 & 24..31
    union { v16h v; u32x4 q[2]; } af;
    const _Float16* ap = A + (size_t)arow * K + kk * 32 + lhi * 8;
    af.q[0] = *(const u32x4*)(ap);
    af.q[1] = *(const u32x4*)(ap + 16);

    // B fragment: lane -> col n = n0+16t+l15; lanes 0-15 K 0..7 & 16..23, lanes 16-31 K 8..15 & 24..31
    const _Float16* pb = &panel[cur][0] + l15 * 32 + lhi * 8;
    union { v16h v; u32x4 q[2]; } bf[2];
    {
      const u32x4* bp = (const u32x4*)(pb);
      bf[0].q[0] = bp[0];
      bf[0].q[1] = bp[2];
    }
#pragma unroll
    for (int t = 0; t < 16; ++t) {
      if (t + 1 < 16) {  // preload next tile's fragment before consuming this one
        const u32x4* bp = (const u32x4*)(pb + (t + 1) * 16 * 32);
        bf[(t + 1) & 1].q[0] = bp[0];
        bf[(t + 1) & 1].q[1] = bp[2];
      }
      acc[t] = __builtin_amdgcn_wmma_f32_16x16x32_f16(false, af.v, false, bf[t & 1].v,
                                                      (short)0, acc[t], false, false);
    }
  }

  // C/D layout: lane group lhi -> rows lhi*8..+7, col = l15
#pragma unroll
  for (int t = 0; t < 16; ++t)
#pragma unroll
    for (int v = 0; v < 8; ++v) {
      int r = m0 + lhi * 8 + v;
      if (r < M) C[(size_t)r * Nstride + n0 + t * 16 + l15] = acc[t][v];
    }
}

// ---------- pooling ----------
__global__ __launch_bounds__(256) void k_pool_node(const _Float16* __restrict__ h16,
                                                   const int* __restrict__ batch,
                                                   float* __restrict__ gsum,
                                                   unsigned* __restrict__ gmax,
                                                   float* __restrict__ cnt, int n4) {
  int t = blockIdx.x * 256 + threadIdx.x;
  if (t >= n4) return;
  int i = t >> 6, g = t & 63;
  int gr = batch[i];
  const _Float16* hp = h16 + (size_t)i * EMB + g * 4;
#pragma unroll
  for (int j = 0; j < 4; ++j) {
    float v = (float)hp[j];                    // >= 0 after ReLU
    atomicAdd(&gsum[(size_t)gr * EMB + g * 4 + j], v);
    atomicMax(&gmax[(size_t)gr * EMB + g * 4 + j], __float_as_uint(v));
  }
  if (g == 0) atomicAdd(&cnt[gr], 1.0f);
}

// hid_in16[g][0..255]=gmax, [256..511]=gsum/max(cnt,1)
__global__ __launch_bounds__(256) void k_pool_concat(const unsigned* __restrict__ gmax,
                                                     const float* __restrict__ gsum,
                                                     const float* __restrict__ cnt,
                                                     _Float16* __restrict__ hid16) {
  int t = blockIdx.x * 256 + threadIdx.x;
  if (t >= NGRAPH * 512) return;
  int g = t >> 9, f = t & 511;
  float v;
  if (f < EMB) v = __uint_as_float(gmax[(size_t)g * EMB + f]);
  else         v = gsum[(size_t)g * EMB + (f - EMB)] / fmaxf(cnt[g], 1.0f);
  hid16[t] = (_Float16)v;
}

// hidden = relu(C + b); write f32 in place + f16 copy for the next GEMM
__global__ __launch_bounds__(256) void k_bias_relu(float* __restrict__ C,
                                                   const float* __restrict__ bias,
                                                   _Float16* __restrict__ out16,
                                                   int total, int N) {
  int t = blockIdx.x * 256 + threadIdx.x;
  if (t >= total) return;
  int f = t % N;
  float v = fmaxf(C[t] + bias[f], 0.0f);
  C[t] = v;
  out16[t] = (_Float16)v;
}

// out[g] = hidden[g] . Ow + Ob  (one wave32 per graph)
__global__ __launch_bounds__(256) void k_final_dot(const float* __restrict__ hidden,
                                                   const float* __restrict__ Ow,
                                                   const float* __restrict__ Ob,
                                                   float* __restrict__ out) {
  int wave = threadIdx.x >> 5, lane = threadIdx.x & 31;
  int g = blockIdx.x * 8 + wave;
  const float* hp = hidden + (size_t)g * 512;
  float s = 0.0f;
#pragma unroll
  for (int j = 0; j < 16; ++j) s = fmaf(hp[lane + 32 * j], Ow[lane + 32 * j], s);
#pragma unroll
  for (int off = 16; off; off >>= 1) s += __shfl_xor(s, off, 32);
  if (lane == 0) out[g] = s + Ob[0];
}

// ---------- host ----------
static inline int blks(long long n) { return (int)((n + 255) / 256); }

extern "C" void kernel_launch(void* const* d_in, const int* in_sizes, int n_in,
                              void* d_out, int out_size, void* d_ws, size_t ws_size,
                              hipStream_t stream) {
  const float* x     = (const float*)d_in[0];
  const int*   eidx  = (const int*)d_in[1];
  const int*   batch = (const int*)d_in[2];
  const float* W0 = (const float*)d_in[3];  const float* b0 = (const float*)d_in[4];
  const float* W1 = (const float*)d_in[5];  const float* b1 = (const float*)d_in[6];
  const float* W2 = (const float*)d_in[7];  const float* b2 = (const float*)d_in[8];
  const float* W3 = (const float*)d_in[9];  const float* b3 = (const float*)d_in[10];
  const float* Lw0 = (const float*)d_in[11]; const float* Lb0 = (const float*)d_in[12];
  const float* Lw1 = (const float*)d_in[13]; const float* Lb1 = (const float*)d_in[14];
  const float* Ow  = (const float*)d_in[15]; const float* Ob  = (const float*)d_in[16];
  float* out = (float*)d_out;

  const int* src = eidx;
  const int* dst = eidx + NEDGES;

  // workspace carve-up
  char* p = (char*)d_ws;
  auto carve = [&](size_t bytes) { char* r = p; p += (bytes + 255) & ~(size_t)255; return r; };
  float*    dinv    = (float*)carve((size_t)NNODES * 4);
  float*    hW      = (float*)carve((size_t)NNODES * EMB * 4);
  float*    agg     = (float*)carve((size_t)NNODES * EMB * 4);
  _Float16* h16     = (_Float16*)carve((size_t)NNODES * EMB * 2);
  _Float16* Wt16    = (_Float16*)carve((size_t)512 * 512 * 2);
  unsigned* gmax    = (unsigned*)carve((size_t)NGRAPH * EMB * 4);
  float*    gsum    = (float*)carve((size_t)NGRAPH * EMB * 4);
  float*    cnt     = (float*)carve((size_t)NGRAPH * 4);
  _Float16* hidA16  = (_Float16*)carve((size_t)NGRAPH * 512 * 2);
  float*    hid0    = (float*)carve((size_t)NGRAPH * 512 * 4);
  _Float16* hid016  = (_Float16*)carve((size_t)NGRAPH * 512 * 2);

  float* hidden_out = out + NGRAPH;  // [1024][512] region of d_out

  const int n4  = NNODES * 64;   // node*float4 threads
  const int e64 = NEDGES * 64;

  // degree -> dinv
  k_zero_f32<<<blks(NNODES), 256, 0, stream>>>(dinv, NNODES);
  k_deg_scatter<<<blks(NEDGES), 256, 0, stream>>>(dst, dinv, NEDGES);
  k_deg_to_dinv<<<blks(NNODES), 256, 0, stream>>>(dinv, NNODES);

  // layer 0 (K=9 scalar GEMM)
  k_gemm0<<<NNODES, 256, 0, stream>>>(x, W0, hW);
  k_agg_init<<<blks(n4), 256, 0, stream>>>(hW, dinv, b0, agg, n4);
  k_edge_scatter<<<blks(e64), 256, 0, stream>>>(src, dst, hW, dinv, agg, e64);
  k_relu_cvt<<<blks(n4), 256, 0, stream>>>(agg, h16, n4);

  // layers 1..3 (WMMA GEMM 100000x256x256)
  const float* Ws[3] = {W1, W2, W3};
  const float* bs[3] = {b1, b2, b3};
  dim3 ggrid((NNODES + 127) / 128, 1);
  for (int l = 0; l < 3; ++l) {
    k_wcvt_t<<<blks(EMB * EMB), 256, 0, stream>>>(Ws[l], Wt16, EMB, EMB);
    k_gemm_wmma<<<ggrid, 256, 0, stream>>>(h16, Wt16, hW, NNODES, EMB, EMB);
    k_agg_init<<<blks(n4), 256, 0, stream>>>(hW, dinv, bs[l], agg, n4);
    k_edge_scatter<<<blks(e64), 256, 0, stream>>>(src, dst, hW, dinv, agg, e64);
    k_relu_cvt<<<blks(n4), 256, 0, stream>>>(agg, h16, n4);
  }

  // pooling
  k_zero_f32<<<blks(NGRAPH * EMB), 256, 0, stream>>>((float*)gmax, NGRAPH * EMB); // 0 bits == 0.0f
  k_zero_f32<<<blks(NGRAPH * EMB), 256, 0, stream>>>(gsum, NGRAPH * EMB);
  k_zero_f32<<<blks(NGRAPH), 256, 0, stream>>>(cnt, NGRAPH);
  k_pool_node<<<blks(n4), 256, 0, stream>>>(h16, batch, gsum, gmax, cnt, n4);
  k_pool_concat<<<blks(NGRAPH * 512), 256, 0, stream>>>(gmax, gsum, cnt, hidA16);

  // MLP head: 1024x512x512 WMMA GEMMs
  dim3 mgrid(NGRAPH / 128, 512 / 256);
  k_wcvt_t<<<blks(512 * 512), 256, 0, stream>>>(Lw0, Wt16, 512, 512);
  k_gemm_wmma<<<mgrid, 256, 0, stream>>>(hidA16, Wt16, hid0, NGRAPH, 512, 512);
  k_bias_relu<<<blks(NGRAPH * 512), 256, 0, stream>>>(hid0, Lb0, hid016, NGRAPH * 512, 512);

  k_wcvt_t<<<blks(512 * 512), 256, 0, stream>>>(Lw1, Wt16, 512, 512);
  k_gemm_wmma<<<mgrid, 256, 0, stream>>>(hid016, Wt16, hidden_out, NGRAPH, 512, 512);
  k_bias_relu<<<blks(NGRAPH * 512), 256, 0, stream>>>(hidden_out, Lb1, hidA16, NGRAPH * 512, 512);

  // out[g] = hidden . Ow + Ob
  k_final_dot<<<NGRAPH / 8, 256, 0, stream>>>(hidden_out, Ow, Ob, out);
}
